// RNN_Model_22333829939560
// MI455X (gfx1250) — matrix operand order repaired and checked by reading, not compiled, not run on hardware
//
#include <hip/hip_runtime.h>
#include <hip/hip_bf16.h>
#include <math.h>

// Problem dims (fixed by reference)
#define BB  256
#define TT  512
#define DD  256
#define HH  512
#define FF1 128
#define OO  64
#define BT  (BB * TT)

typedef __attribute__((ext_vector_type(16))) __bf16 v16bf;
typedef __attribute__((ext_vector_type(8)))  float  v8f;

union FragB16 { v16bf v; unsigned u[8]; uint4 q[2]; };

__device__ __forceinline__ unsigned short f2bf(float f) {
  unsigned u = __builtin_bit_cast(unsigned, f);
  u += 0x7FFFu + ((u >> 16) & 1u);           // round-to-nearest-even
  return (unsigned short)(u >> 16);
}
__device__ __forceinline__ float bf2f(unsigned short h) {
  return __builtin_bit_cast(float, ((unsigned)h) << 16);
}
__device__ __forceinline__ float bflo(unsigned u) {          // bf16 in low 16 bits
  return __builtin_bit_cast(float, u << 16);
}
__device__ __forceinline__ float bfhi(unsigned u) {          // bf16 in high 16 bits
  return __builtin_bit_cast(float, u & 0xffff0000u);
}
__device__ __forceinline__ unsigned pack2bf(float lo, float hi) {
  return (unsigned)f2bf(lo) | ((unsigned)f2bf(hi) << 16);
}

// Native CDNA5 V_TANH_F32 if the toolchain exposes it; ocml fallback otherwise.
#if __has_builtin(__builtin_amdgcn_tanhf)
__device__ __forceinline__ float fast_tanh(float x) { return __builtin_amdgcn_tanhf(x); }
#else
__device__ __forceinline__ float fast_tanh(float x) { return tanhf(x); }
#endif

// ---------------------------------------------------------------- prep
__global__ void k_cvt_bf16(const float* __restrict__ s, unsigned short* __restrict__ d, int n) {
  int i = blockIdx.x * 256 + threadIdx.x;
  if (i < n) d[i] = f2bf(s[i]);
}
__global__ void k_bias_sum(const float* __restrict__ a, const float* __restrict__ b,
                           float* __restrict__ o, int n) {
  int i = blockIdx.x * 256 + threadIdx.x;
  if (i < n) o[i] = a[i] + b[i];
}

// ---------------------------------------------------------------- pre-GEMM
// pre[m, n] = sum_k A[m,k] * W[n,k] + bias[n]   (M = B*T, N = H, K = DD or HH)
// Output is written in the recurrence-fragment-friendly permutation:
//   preP[ (b/16)*T + t ][ n*16 + (b%16) ]   (each 16x512 step tile = contiguous 16 KB)
// Block: 256 thr (8 waves). Wave w: M-strip (w&1), N-range (w>>1)*128 (8 wmma tiles).
template <bool ABF16>
__global__ __launch_bounds__(256) void k_gemm_pre(
    const void* __restrict__ Aptr, const unsigned short* __restrict__ Wbf,
    const float* __restrict__ bias, unsigned short* __restrict__ Cbf, int K)
{
  const int lane = threadIdx.x & 31;
  const int w    = threadIdx.x >> 5;
  const int half = lane >> 4;
  const int l16  = lane & 15;
  const long mBase = (long)blockIdx.x * 32 + (w & 1) * 16;
  const int  nBase = (w >> 1) * 128;
  const long aRow  = mBase + l16;

  v8f acc[8] = {};
  for (int kb = 0; kb < K; kb += 32) {
    // A fragment: lane = row M, half selects K sub-blocks {kb+half*8 .. +7} and {+16..}
    FragB16 a;
    if constexpr (ABF16) {
      const uint4* ap = (const uint4*)((const unsigned short*)Aptr + aRow * K + kb + half * 8);
      a.q[0] = ap[0]; a.q[1] = ap[2];
    } else {
      const float*  af = (const float*)Aptr + aRow * K + kb + half * 8;
      const float4* f0 = (const float4*)af;
      const float4* f1 = (const float4*)(af + 16);
      float4 c0 = f0[0], c1 = f0[1], c2 = f1[0], c3 = f1[1];
      a.u[0] = pack2bf(c0.x, c0.y); a.u[1] = pack2bf(c0.z, c0.w);
      a.u[2] = pack2bf(c1.x, c1.y); a.u[3] = pack2bf(c1.z, c1.w);
      a.u[4] = pack2bf(c2.x, c2.y); a.u[5] = pack2bf(c2.z, c2.w);
      a.u[6] = pack2bf(c3.x, c3.y); a.u[7] = pack2bf(c3.z, c3.w);
    }
#pragma unroll
    for (int nt = 0; nt < 8; nt++) {
      const int n = nBase + nt * 16 + l16;          // lane = column N
      const uint4* bp = (const uint4*)(Wbf + (long)n * K + kb + half * 8);
      FragB16 b; b.q[0] = bp[0]; b.q[1] = bp[2];    // K sub-blocks {half*8..+7} and {+16..}
      acc[nt] = __builtin_amdgcn_wmma_f32_16x16x32_bf16(
          false, a.v, false, b.v, (short)0, acc[nt], false, false);
    }
  }
#pragma unroll
  for (int nt = 0; nt < 8; nt++) {
    const int n  = nBase + nt * 16 + l16;
    const float bv = bias[n];
#pragma unroll
    for (int i = 0; i < 8; i++) {
      const long m = mBase + i + 8 * half;          // global row = b*T + t
      const int  b = (int)(m >> 9);                 // TT == 512
      const int  t = (int)(m & (TT - 1));
      const size_t dst = ((size_t)(b >> 4) * TT + t) * (16 * HH) + (size_t)n * 16 + (b & 15);
      Cbf[dst] = f2bf(acc[nt][i] + bv);
    }
  }
}

// ---------------------------------------------------------------- recurrence
// One WG owns 16 batch rows for all T steps. h tile lives in LDS (bf16).
// pre_t tiles (16 KB bf16, permuted layout) double-buffered via async-to-LDS loads.
__global__ __launch_bounds__(256) void k_rnn_layer(
    const unsigned short* __restrict__ prePerm,  // permuted pre (+biases), bf16
    const unsigned short* __restrict__ Whh,      // [H, H] bf16 row-major
    unsigned short* __restrict__ hsOut,          // [B*T, H] bf16 or nullptr
    float* __restrict__ lastOut)                 // [B, H] f32 or nullptr
{
  // ushort layout: hT [0,8192)  pb0 [8192,16384)  pb1 [16384,24576)   = 48 KB
  __shared__ __align__(16) unsigned short smem[3 * 16 * HH];

  const int tid  = threadIdx.x;
  const int lane = tid & 31;
  const int w    = tid >> 5;
  const int half = lane >> 4;
  const int l16  = lane & 15;
  const int b0   = blockIdx.x * 16;     // batch rows [b0, b0+16)
  const int nBase = w * 64;             // wave covers 4 N-tiles

  const unsigned long long preAddr = (unsigned long long)prePerm;
  const unsigned smemBase = (unsigned)(unsigned long long)(&smem[0]); // low 32b = LDS offset

  // Async copy of the contiguous 16 KB permuted pre tile for step t into buffer buf.
  // 1024 x b128 per workgroup = 4 per thread; each async op moves 32 lanes x 16B.
  auto issuePre = [&](int t, int buf) {
    const unsigned ldsB = smemBase + (8192u + (unsigned)buf * 8192u) * 2u;
    const long tileBase = ((long)blockIdx.x * TT + t) * (16 * HH);   // element index
#pragma unroll
    for (int c = 0; c < 4; c++) {
      const int e = (c * 256 + tid) * 8;           // chunk of 8 ushorts = 16B
      const unsigned lds  = ldsB + (unsigned)e * 2u;
      const unsigned goff = (unsigned)((tileBase + e) * 2);
      asm volatile("global_load_async_to_lds_b128 %0, %1, %2 offset:0"
                   :: "v"(lds), "v"(goff), "s"(preAddr) : "memory");
    }
  };

  issuePre(0, 0);

  for (int t = 0; t < TT; t++) {
    const int cur = t & 1;
    if (t + 1 < TT) {
      issuePre(t + 1, cur ^ 1);
      asm volatile("s_wait_asynccnt 4" ::: "memory");  // step-t chunks (in-order) done
    } else {
      asm volatile("s_wait_asynccnt 0" ::: "memory");
    }
    __syncthreads();  // pre[t] visible to all waves; h writes of t-1 visible

    v8f acc[4] = {};
    if (t > 0) {
      for (int kb = 0; kb < HH; kb += 32) {
        FragB16 a;  // A = h tile from LDS, lane = row M
        const uint4* ap = (const uint4*)&smem[l16 * HH + kb + half * 8];
        a.q[0] = ap[0]; a.q[1] = ap[2];
#pragma unroll
        for (int nt = 0; nt < 4; nt++) {
          const int n = nBase + nt * 16 + l16;
          const uint4* bp = (const uint4*)(Whh + (long)n * HH + kb + half * 8);
          FragB16 b; b.q[0] = bp[0]; b.q[1] = bp[2];
          acc[nt] = __builtin_amdgcn_wmma_f32_16x16x32_bf16(
              false, a.v, false, b.v, (short)0, acc[nt], false, false);
        }
      }
    }

    // h_new = tanh(pre_t + acc); pre tile is in C-fragment order: one b128 per N-tile.
    float hnew[4][8];
    const unsigned short* pbuf = &smem[8192 + cur * 8192];
#pragma unroll
    for (int nt = 0; nt < 4; nt++) {
      const int n = nBase + nt * 16 + l16;
      const uint4 pv = *(const uint4*)&pbuf[(unsigned)n * 16 + half * 8];
      hnew[nt][0] = fast_tanh(bflo(pv.x) + acc[nt][0]);
      hnew[nt][1] = fast_tanh(bfhi(pv.x) + acc[nt][1]);
      hnew[nt][2] = fast_tanh(bflo(pv.y) + acc[nt][2]);
      hnew[nt][3] = fast_tanh(bfhi(pv.y) + acc[nt][3]);
      hnew[nt][4] = fast_tanh(bflo(pv.z) + acc[nt][4]);
      hnew[nt][5] = fast_tanh(bfhi(pv.z) + acc[nt][5]);
      hnew[nt][6] = fast_tanh(bflo(pv.w) + acc[nt][6]);
      hnew[nt][7] = fast_tanh(bfhi(pv.w) + acc[nt][7]);
    }
    __syncthreads();  // all reads of old h tile complete before overwrite

#pragma unroll
    for (int nt = 0; nt < 4; nt++) {
      const int n = nBase + nt * 16 + l16;
#pragma unroll
      for (int i = 0; i < 8; i++) {
        const int m = i + 8 * half;
        smem[m * HH + n] = f2bf(hnew[nt][i]);
        if (lastOut && t == TT - 1) lastOut[(long)(b0 + m) * HH + n] = hnew[nt][i];
      }
    }

    if (hsOut) {
      __syncthreads();  // h tile fully written -> coalesced b128 copy to global
#pragma unroll
      for (int c = 0; c < 4; c++) {
        const int e   = (c * 256 + tid) * 8;
        const int row = e >> 9;
        const int col = e & 511;
        const uint4 vq = *(const uint4*)&smem[e];
        *(uint4*)&hsOut[((long)(b0 + row) * TT + t) * HH + col] = vq;
      }
    }
  }
}

// ---------------------------------------------------------------- head (f32, tiny)
__global__ void k_head1(const float* __restrict__ lastH, const float* __restrict__ Wf1,
                        const float* __restrict__ bf1, float* __restrict__ z) {
  const int idx = blockIdx.x * 256 + threadIdx.x;   // B*F1
  const int b = idx / FF1, f = idx % FF1;
  const float* hp = lastH + (long)b * HH;
  const float* wp = Wf1 + (long)f * HH;
  float s = 0.f;
  for (int k = 0; k < HH; k += 4) {
    float4 hv = *(const float4*)(hp + k);
    float4 wv = *(const float4*)(wp + k);
    s += hv.x * wv.x + hv.y * wv.y + hv.z * wv.z + hv.w * wv.w;
  }
  s += bf1[f];
  z[idx] = s > 0.f ? s : 0.f;
}
__global__ void k_head2(const float* __restrict__ z, const float* __restrict__ Wf2,
                        const float* __restrict__ bf2, float* __restrict__ out) {
  const int idx = blockIdx.x * 256 + threadIdx.x;   // B*O
  const int b = idx / OO, o = idx % OO;
  const float* zp = z + (long)b * FF1;
  const float* wp = Wf2 + (long)o * FF1;
  float s = 0.f;
  for (int k = 0; k < FF1; k += 4) {
    float4 zv = *(const float4*)(zp + k);
    float4 wv = *(const float4*)(wp + k);
    s += zv.x * wv.x + zv.y * wv.y + zv.z * wv.z + zv.w * wv.w;
  }
  out[idx] = s + bf2[o];
}

// ---------------------------------------------------------------- launch
extern "C" void kernel_launch(void* const* d_in, const int* in_sizes, int n_in,
                              void* d_out, int out_size, void* d_ws, size_t ws_size,
                              hipStream_t stream) {
  (void)in_sizes; (void)n_in; (void)out_size;
  const float* x    = (const float*)d_in[0];
  const float* Wih0 = (const float*)d_in[1];
  const float* Whh0 = (const float*)d_in[2];
  const float* bih0 = (const float*)d_in[3];
  const float* bhh0 = (const float*)d_in[4];
  const float* Wih1 = (const float*)d_in[5];
  const float* Whh1 = (const float*)d_in[6];
  const float* bih1 = (const float*)d_in[7];
  const float* bhh1 = (const float*)d_in[8];
  const float* Wf1  = (const float*)d_in[9];
  const float* bf1  = (const float*)d_in[10];
  const float* Wf2  = (const float*)d_in[11];
  const float* bf2  = (const float*)d_in[12];
  float* out = (float*)d_out;

  char* ws = (char*)d_ws;
  size_t off = 0;
  auto carve = [&](size_t bytes) { void* p = ws + off; off = (off + bytes + 255) & ~(size_t)255; return p; };
  unsigned short* wih0b = (unsigned short*)carve((size_t)HH * DD * 2);
  unsigned short* whh0b = (unsigned short*)carve((size_t)HH * HH * 2);
  unsigned short* wih1b = (unsigned short*)carve((size_t)HH * HH * 2);
  unsigned short* whh1b = (unsigned short*)carve((size_t)HH * HH * 2);
  float*          b0    = (float*)carve((size_t)HH * 4);
  float*          b1    = (float*)carve((size_t)HH * 4);
  float*          lasth = (float*)carve((size_t)BB * HH * 4);
  float*          zbuf  = (float*)carve((size_t)BB * FF1 * 4);
  unsigned short* pre   = (unsigned short*)carve((size_t)BT * HH * 2);
  unsigned short* hs0   = (unsigned short*)carve((size_t)BT * HH * 2);
  if (off > ws_size) return;  // workspace too small; refuse to scribble

  // weight prep (f32 -> bf16) and bias folding
  k_cvt_bf16<<<(HH * DD + 255) / 256, 256, 0, stream>>>(Wih0, wih0b, HH * DD);
  k_cvt_bf16<<<(HH * HH + 255) / 256, 256, 0, stream>>>(Whh0, whh0b, HH * HH);
  k_cvt_bf16<<<(HH * HH + 255) / 256, 256, 0, stream>>>(Wih1, wih1b, HH * HH);
  k_cvt_bf16<<<(HH * HH + 255) / 256, 256, 0, stream>>>(Whh1, whh1b, HH * HH);
  k_bias_sum<<<(HH + 255) / 256, 256, 0, stream>>>(bih0, bhh0, b0, HH);
  k_bias_sum<<<(HH + 255) / 256, 256, 0, stream>>>(bih1, bhh1, b1, HH);

  // layer 0
  k_gemm_pre<false><<<BT / 32, 256, 0, stream>>>(x, wih0b, b0, pre, DD);
  k_rnn_layer<<<BB / 16, 256, 0, stream>>>(pre, whh0b, hs0, nullptr);
  // layer 1
  k_gemm_pre<true><<<BT / 32, 256, 0, stream>>>(hs0, wih1b, b1, pre, HH);
  k_rnn_layer<<<BB / 16, 256, 0, stream>>>(pre, whh1b, nullptr, lasth);
  // head
  k_head1<<<(BB * FF1) / 256, 256, 0, stream>>>(lasth, Wf1, bf1, zbuf);
  k_head2<<<(BB * OO) / 256, 256, 0, stream>>>(zbuf, Wf2, bf2, out);
}
